// ZcCBAM_1872605741352
// MI455X (gfx1250) — compile-verified, matrix operand order checked by reading, not run on hardware
//
#include <hip/hip_runtime.h>
#include <hip/hip_bf16.h>
#include <math.h>
#include <stdint.h>

typedef __bf16 bf16;
typedef __bf16  v16bf __attribute__((ext_vector_type(16)));
typedef float   v8f   __attribute__((ext_vector_type(8)));
typedef unsigned int v4u __attribute__((ext_vector_type(4)));
typedef int          v8i __attribute__((ext_vector_type(8)));
typedef int          v4i __attribute__((ext_vector_type(4)));

#define NUM_BINS 8
#define C_OUT    64
#define HID1     32          // C_OUT/2
#define K1       512         // C_OUT*NUM_BINS
#define N1       256         // K1/2
#define N2       64
#define NKT1     16          // K1/32
#define NCT1     16          // N1/16
#define NKT2     8           // N1/32
#define NCT2     4           // N2/16

#if defined(__has_builtin)
#  if __has_builtin(__builtin_amdgcn_tensor_load_to_lds)
#    define HAVE_TDM 1
#  endif
#endif

__device__ __forceinline__ float sigmoidf_(float x) { return 1.0f / (1.0f + expf(-x)); }

// ISA 16-bit A-matrix 16x32 fragment layout (wave32):
//  lane<16  : row m=lane,   e<8 -> K=e,    e>=8 -> K=e+8
//  lane>=16 : row m=lane-16,e<8 -> K=e+8,  e>=8 -> K=e+16
__device__ __forceinline__ void frag_pos(int r, int m, int& lane, int& e) {
    if (r < 8)       { lane = m;      e = r;      }
    else if (r < 16) { lane = m + 16; e = r - 8;  }
    else if (r < 24) { lane = m;      e = r - 8;  }
    else             { lane = m + 16; e = r - 16; }
}
__device__ __forceinline__ int frag_koff(int lane, int e) {
    return (lane < 16) ? ((e < 8) ? e : e + 8) : ((e < 8) ? e + 8 : e + 16);
}

#ifdef HAVE_TDM
// TDM D# (CDNA5 ISA): 1-D tile, 8-byte elements,
// 2048 elements = 16 KB contiguous global -> LDS.
__device__ __forceinline__ void tdm_load_tile_16k(const bf16* gsrc, void* lds_dst) {
    unsigned long long ga = (unsigned long long)(uintptr_t)gsrc;
    unsigned int lds = (unsigned int)(uintptr_t)lds_dst;   // low 32 bits = LDS byte offset
    v4u g0 = { 1u,                                          // count=1, user descriptor
               lds,                                         // lds_addr [63:32]
               (unsigned int)(ga & 0xFFFFFFFFu),            // global_addr lo
               (unsigned int)(((ga >> 32) & 0x01FFFFFFu)    // global_addr hi (57-bit)
                              | (2u << 30)) };              // type=2 ("image")
    v8i g1 = { (int)(3u << 16),                             // data_size=3 (8B), wg_mask=0
               (int)((2048u & 0xFFFFu) << 16),              // tensor_dim0[15:0] @ [31:16]
               0,                                           // tensor_dim0 hi / tensor_dim1 lo
               (int)((2048u & 0xFFFFu) << 16),              // tile_dim0=2048 @ [127:112]
               0,                                           // tile_dim1=0 (unused), tile_dim2=0
               2048,                                        // tensor_dim0_stride lo32
               0, 0 };
    v4i gz = { 0, 0, 0, 0 };
#if __has_include(<hip/amd_detail/amd_gfx1250_TDM.h>)
    v8i gz8 = { 0, 0, 0, 0, 0, 0, 0, 0 };
    __builtin_amdgcn_tensor_load_to_lds(g0, g1, gz, gz, gz8, 0);   // clang-23 6-arg form
#else
    __builtin_amdgcn_tensor_load_to_lds(g0, g1, gz, gz, 0);        // ROCm 7.2 5-arg form
#endif
}
#endif

// ---------------------------------------------------------------- utilities
__global__ void k_zero_f32(float* p, long long n) {
    long long i = (long long)blockIdx.x * blockDim.x + threadIdx.x;
    long long stride = (long long)gridDim.x * blockDim.x;
    for (; i < n; i += stride) p[i] = 0.0f;
}

__global__ void k_mask(const int* cnt, int* pos, int U) {
    int u = blockIdx.x * blockDim.x + threadIdx.x;
    if (u < U) pos[u] = (cnt[u] >= 2) ? 1 : 0;
}

__global__ void k_scan1(int* pos, int* tmp, int U) {
    __shared__ int s[256];
    int t = threadIdx.x;
    int base = blockIdx.x * 1024 + t * 4;
    int v[4]; int loc = 0;
#pragma unroll
    for (int j = 0; j < 4; j++) { int u = base + j; v[j] = (u < U) ? pos[u] : 0; loc += v[j]; }
    s[t] = loc; __syncthreads();
    for (int off = 1; off < 256; off <<= 1) {
        int x = (t >= off) ? s[t - off] : 0; __syncthreads();
        s[t] += x; __syncthreads();
    }
    int run = s[t] - loc;
#pragma unroll
    for (int j = 0; j < 4; j++) { int u = base + j; if (u < U) { int old = v[j]; pos[u] = run; run += old; } }
    if (t == 255) tmp[blockIdx.x] = s[255];
}

__global__ void k_scan2(int* tmp, int* hdr, int nch) {
    __shared__ int s[256];
    __shared__ int carry;
    int t = threadIdx.x;
    if (t == 0) carry = 0;
    __syncthreads();
    for (int base = 0; base < nch; base += 256) {
        int i = base + t;
        int x = (i < nch) ? tmp[i] : 0;
        s[t] = x; __syncthreads();
        for (int off = 1; off < 256; off <<= 1) {
            int y = (t >= off) ? s[t - off] : 0; __syncthreads();
            s[t] += y; __syncthreads();
        }
        if (i < nch) tmp[i] = carry + s[t] - x;
        __syncthreads();
        if (t == 255) carry += s[255];
        __syncthreads();
    }
    if (t == 0) hdr[0] = carry;   // M
}

__global__ void k_scan3(int* pos, const int* tmp, int U) {
    int u = blockIdx.x * blockDim.x + threadIdx.x;
    if (u < U) pos[u] += tmp[u >> 10];
}

__global__ void k_scatter(const float* vf, const int* coords, const int* inv,
                          float* feat, int Nv) {
    int v = blockIdx.x * blockDim.x + threadIdx.x;
    if (v >= Nv) return;
    int u = inv[v];
    int b = coords[v * 4 + 1];
    float* dst = feat + ((size_t)u * NUM_BINS + b) * 5;
#pragma unroll
    for (int j = 0; j < 5; j++) dst[j] = vf[v * 5 + j];
}

// pack bs_w1 / bs_w2 into per-lane-contiguous B-fragment bf16 layout
__global__ void k_pack_w(const float* w1, const float* w2, bf16* w1p, bf16* w2p) {
    int a = blockIdx.x * blockDim.x + threadIdx.x;
    const int tot1 = K1 * N1;
    const int tot2 = N1 * N2;
    if (a < tot1) {
        int e = a & 15, lane = (a >> 4) & 31, rest = a >> 9;
        int ct = rest % NCT1, kt = rest / NCT1;
        int n = ct * 16 + (lane & 15);
        int k = kt * 32 + frag_koff(lane, e);
        w1p[a] = (bf16)w1[k * N1 + n];
    } else if (a < tot1 + tot2) {
        int a2 = a - tot1;
        int e = a2 & 15, lane = (a2 >> 4) & 31, rest = a2 >> 9;
        int ct = rest % NCT2, kt = rest / NCT2;
        int n = ct * 16 + (lane & 15);
        int k = kt * 32 + frag_koff(lane, e);
        w2p[a2] = (bf16)w2[k * N2 + n];
    }
}

// ------------------------------------------------- per-row MLP + attention
__global__ void k_stage_a(const int* cnt, const int* pos, const float* feat,
                          const float* up_w1, const float* up_b1,
                          const float* up_w2, const float* up_b2,
                          const float* ca_w1, const float* ca_w2,
                          const float* sa_conv, bf16* flatA, int U) {
    int u = blockIdx.x;
    if (u >= U || cnt[u] < 2) return;
    int p = pos[u];
    int t = threadIdx.x;                    // 64 threads
    __shared__ float fL[NUM_BINS * 5];
    __shared__ float H[NUM_BINS * HID1];
    __shared__ float X[C_OUT * NUM_BINS];   // channel-major [c][b]
    __shared__ float mean_[C_OUT], mx_[C_OUT];
    __shared__ float hm[8], hx[8], bm[8], bx[8], sa[8];

    if (t < 40) fL[t] = feat[(size_t)u * 40 + t];
    __syncthreads();

    // hidden = relu(feat @ up_w1 + b1) : (8,32)
#pragma unroll
    for (int j = 0; j < 4; j++) {
        int idx = t * 4 + j; int b = idx >> 5; int h = idx & 31;
        float s = up_b1[h];
#pragma unroll
        for (int i = 0; i < 5; i++) s += fL[b * 5 + i] * up_w1[i * HID1 + h];
        H[idx] = fmaxf(s, 0.0f);
    }
    __syncthreads();

    // h = hidden @ up_w2 + b2 -> store transposed X[c][b]
#pragma unroll
    for (int j = 0; j < 8; j++) {
        int idx = t * 8 + j; int b = idx >> 6; int c = idx & 63;
        float s = up_b2[c];
        for (int h = 0; h < HID1; h++) s += H[b * HID1 + h] * up_w2[h * C_OUT + c];
        X[c * NUM_BINS + b] = s;
    }
    __syncthreads();

    { // per-channel mean / max over bins
        int c = t; float s = 0.0f, m = -3.4e38f;
#pragma unroll
        for (int b = 0; b < 8; b++) { float v = X[c * 8 + b]; s += v; m = fmaxf(m, v); }
        mean_[c] = s * 0.125f; mx_[c] = m;
    }
    __syncthreads();
    if (t < 16) { // ca_mlp hidden (8) for mean (t<8) and max (t>=8)
        int r = t & 7; const float* src = (t < 8) ? mean_ : mx_;
        float s = 0.0f;
        for (int c = 0; c < 64; c++) s += src[c] * ca_w1[c * 8 + r];
        float v = fmaxf(s, 0.0f);
        if (t < 8) hm[r] = v; else hx[r] = v;
    }
    __syncthreads();
    { // ca = sigmoid(hm@W2 + hx@W2) ; scale channels
        int c = t; float a = 0.0f;
#pragma unroll
        for (int r = 0; r < 8; r++) a += (hm[r] + hx[r]) * ca_w2[r * 64 + c];
        float cav = sigmoidf_(a);
#pragma unroll
        for (int b = 0; b < 8; b++) X[c * 8 + b] *= cav;
    }
    __syncthreads();
    if (t < 8) { // per-bin mean/max over channels
        int b = t; float s = 0.0f, m = -3.4e38f;
        for (int c = 0; c < 64; c++) { float v = X[c * 8 + b]; s += v; m = fmaxf(m, v); }
        bm[b] = s * (1.0f / 64.0f); bx[b] = m;
    }
    __syncthreads();
    if (t < 8) { // 1D conv, width 7, pad 3, 2 input channels
        int b = t; float s = 0.0f;
#pragma unroll
        for (int k = 0; k < 7; k++) {
            int q = b + k - 3;
            if (q >= 0 && q < 8) s += bm[q] * sa_conv[k] + bx[q] * sa_conv[7 + k];
        }
        sa[b] = sigmoidf_(s);
    }
    __syncthreads();

    // scale by sa and emit flat row (index k = c*8+b) in WMMA-A-packed bf16 layout
    int rt = p >> 4, m = p & 15;
    bf16* base = flatA + (size_t)rt * NKT1 * 32 * 16;
#pragma unroll
    for (int j = 0; j < 8; j++) {
        int idx = t * 8 + j; int c = idx >> 3; int b = idx & 7;
        float v = X[c * 8 + b] * sa[b];
        int k = idx;                     // c*8 + b
        int kt = k >> 5, r = k & 31;
        int lane, e; frag_pos(r, m, lane, e);
        base[((size_t)kt * 32 + lane) * 16 + e] = (bf16)v;
    }
}

// --------------------------------------------------- GEMM1 (Mx512 @ 512x256)
__global__ void __launch_bounds__(256, 2)
k_gemm1(const bf16* __restrict__ flatA, const bf16* __restrict__ w1p,
        bf16* __restrict__ z1row, float* sums1, const int* hdr, int U) {
    int M = hdr[0];
    int p0 = blockIdx.x * 16;
    if (p0 >= M) return;                       // block-uniform exit
    int t = threadIdx.x, wv = t >> 5, lane = t & 31;

    __shared__ __align__(32) bf16 As[NKT1 * 32 * 16];   // 16 KB A row-tile
#ifdef HAVE_TDM
    // Tensor Data Mover: one DMA of the whole tile, issued by wave 0 only
    if (wv == 0) {
        tdm_load_tile_16k(flatA + (size_t)blockIdx.x * NKT1 * 32 * 16, (void*)As);
        __builtin_amdgcn_s_wait_tensorcnt(0);
    }
#else
    {   // cooperative 16KB load, 64B per thread
        const uint4* src = (const uint4*)(flatA + (size_t)blockIdx.x * NKT1 * 32 * 16);
        uint4* dst = (uint4*)As;
#pragma unroll
        for (int j = 0; j < 4; j++) dst[t + j * 256] = src[t + j * 256];
    }
#endif
    __syncthreads();

    int ct0 = wv * 2, ct1 = wv * 2 + 1;
    v8f acc0 = {}; v8f acc1 = {};
#pragma unroll
    for (int kt = 0; kt < NKT1; kt++) {
        v16bf a = *(const v16bf*)&As[((size_t)kt * 32 + lane) * 16];
        if (kt + 1 < NKT1)
            __builtin_prefetch(w1p + (((size_t)(kt + 1) * NCT1 + ct0) * 32 + lane) * 16, 0, 3);
        v16bf b0 = *(const v16bf*)(w1p + (((size_t)kt * NCT1 + ct0) * 32 + lane) * 16);
        v16bf b1 = *(const v16bf*)(w1p + (((size_t)kt * NCT1 + ct1) * 32 + lane) * 16);
        acc0 = __builtin_amdgcn_wmma_f32_16x16x32_bf16(false, a, false, b0, (short)0, acc0, false, false);
        acc1 = __builtin_amdgcn_wmma_f32_16x16x32_bf16(false, a, false, b1, (short)0, acc1, false, false);
    }

    // D layout: lanes 0-15 N=lane M=v ; lanes 16-31 N=lane-16 M=8+v
    int rowbase = p0 + ((lane >= 16) ? 8 : 0);
    int n0 = ct0 * 16 + (lane & 15), n1 = ct1 * 16 + (lane & 15);
    float s0 = 0, q0 = 0, s1 = 0, q1 = 0;
#pragma unroll
    for (int v = 0; v < 8; v++) {
        int row = rowbase + v;
        float v0 = acc0[v], v1 = acc1[v];
        z1row[(size_t)row * N1 + n0] = (bf16)v0;
        z1row[(size_t)row * N1 + n1] = (bf16)v1;
        if (row < M) { s0 += v0; q0 += v0 * v0; s1 += v1; q1 += v1 * v1; }
    }
    s0 += __shfl_down(s0, 16); q0 += __shfl_down(q0, 16);
    s1 += __shfl_down(s1, 16); q1 += __shfl_down(q1, 16);
    if (lane < 16) {
        atomicAdd(&sums1[n0], s0); atomicAdd(&sums1[N1 + n0], q0);
        atomicAdd(&sums1[n1], s1); atomicAdd(&sums1[N1 + n1], q1);
    }
}

// bn1 + relu, repacked into A-fragment layout for gemm2
__global__ void k_bn1(const bf16* z1row, const float* sums1,
                      const float* g, const float* b,
                      bf16* z1p, const int* hdr, int U) {
    int M = hdr[0];
    int Mc = (M + 15) & ~15;
    int p = blockIdx.x;
    if (p >= Mc) return;
    int c = threadIdx.x;                 // 256
    float invM = 1.0f / (float)M;
    float mu  = sums1[c] * invM;
    float var = sums1[N1 + c] * invM - mu * mu;
    float sc  = rsqrtf(var + 1e-3f) * g[c];
    float y = ((float)z1row[(size_t)p * N1 + c] - mu) * sc + b[c];
    y = fmaxf(y, 0.0f);
    int rt = p >> 4, m = p & 15;
    int kt = c >> 5, r = c & 31;
    int lane, e; frag_pos(r, m, lane, e);
    z1p[(((size_t)rt * NKT2 + kt) * 32 + lane) * 16 + e] = (bf16)y;
}

// --------------------------------------------------- GEMM2 (Mx256 @ 256x64)
__global__ void __launch_bounds__(256, 2)
k_gemm2(const bf16* __restrict__ z1p, const bf16* __restrict__ w2p,
        float* out, float* sums2, const int* hdr, int U) {
    int M = hdr[0];
    int p0b = blockIdx.x * 32;
    if (p0b >= M) return;                      // block-uniform exit
    int t = threadIdx.x, wv = t >> 5, lane = t & 31;
    int rt = (p0b >> 4) + (wv >> 2);           // waves 0-3 tile0, 4-7 tile1
    int ct = wv & 3;
    int p0 = rt * 16;
    v8f acc = {};
#pragma unroll
    for (int kt = 0; kt < NKT2; kt++) {
        v16bf a  = *(const v16bf*)(z1p + (((size_t)rt * NKT2 + kt) * 32 + lane) * 16);
        v16bf bb = *(const v16bf*)(w2p + (((size_t)kt * NCT2 + ct) * 32 + lane) * 16);
        acc = __builtin_amdgcn_wmma_f32_16x16x32_bf16(false, a, false, bb, (short)0, acc, false, false);
    }
    int rowbase = p0 + ((lane >= 16) ? 8 : 0);
    int n = ct * 16 + (lane & 15);
    float s = 0, q = 0;
#pragma unroll
    for (int v = 0; v < 8; v++) {
        int row = rowbase + v;
        float val = acc[v];
        if (row < M) { out[(size_t)row * N2 + n] = val; s += val; q += val * val; }
    }
    s += __shfl_down(s, 16); q += __shfl_down(q, 16);
    if (lane < 16) { atomicAdd(&sums2[n], s); atomicAdd(&sums2[N2 + n], q); }
}

__global__ void k_bn2(float* out, const float* sums2, const float* g, const float* b,
                      const int* hdr) {
    int M = hdr[0];
    int i = blockIdx.x * blockDim.x + threadIdx.x;
    if (i >= M * N2) return;
    int c = i & 63;
    float invM = 1.0f / (float)M;
    float mu  = sums2[c] * invM;
    float var = sums2[N2 + c] * invM - mu * mu;
    float y = (out[i] - mu) * rsqrtf(var + 1e-3f) * g[c] + b[c];
    out[i] = fmaxf(y, 0.0f);
}

// ---------------------------------------------------------------- launcher
extern "C" void kernel_launch(void* const* d_in, const int* in_sizes, int n_in,
                              void* d_out, int out_size, void* d_ws, size_t ws_size,
                              hipStream_t stream) {
    const float* vf    = (const float*)d_in[0];
    const int*   vcrd  = (const int*)  d_in[1];
    const int*   inv   = (const int*)  d_in[3];
    const int*   cnt   = (const int*)  d_in[4];
    const float* up_w1 = (const float*)d_in[5];
    const float* up_b1 = (const float*)d_in[6];
    const float* up_w2 = (const float*)d_in[7];
    const float* up_b2 = (const float*)d_in[8];
    const float* ca_w1 = (const float*)d_in[9];
    const float* ca_w2 = (const float*)d_in[10];
    const float* sa_c  = (const float*)d_in[11];
    const float* bs_w1 = (const float*)d_in[12];
    const float* bn1g  = (const float*)d_in[13];
    const float* bn1b  = (const float*)d_in[14];
    const float* bs_w2 = (const float*)d_in[15];
    const float* bn2g  = (const float*)d_in[16];
    const float* bn2b  = (const float*)d_in[17];

    int Nv  = in_sizes[0] / 5;
    int U   = in_sizes[2];
    int U16 = (U + 15) / 16;
    int Upad = U16 * 16;

    char* ws = (char*)d_ws;
    size_t off = 0;
    auto take = [&](size_t bytes) -> char* {
        char* p = ws + off;
        off = (off + bytes + 255) & ~(size_t)255;
        return p;
    };
    int*   hdr   = (int*)  take(256);
    float* sums1 = (float*)take(2 * N1 * sizeof(float));
    float* sums2 = (float*)take(2 * N2 * sizeof(float));
    int*   stmp  = (int*)  take(4096 * sizeof(int));
    int*   pos   = (int*)  take((size_t)U * sizeof(int));
    float* feat  = (float*)take((size_t)U * 40 * sizeof(float));
    bf16*  w1p   = (bf16*) take((size_t)K1 * N1 * sizeof(bf16));
    bf16*  w2p   = (bf16*) take((size_t)N1 * N2 * sizeof(bf16));
    bf16*  flatA = (bf16*) take((size_t)Upad * K1 * sizeof(bf16));
    bf16*  z1row = (bf16*) take((size_t)Upad * N1 * sizeof(bf16));
    bf16*  z1p   = (bf16*) take((size_t)Upad * N1 * sizeof(bf16));
    (void)ws_size; (void)n_in; (void)out_size;

    // zero counters/stats/feat every call (deterministic across graph replays)
    k_zero_f32<<<1, 256, 0, stream>>>((float*)hdr, 64);
    k_zero_f32<<<2, 256, 0, stream>>>(sums1, 2 * N1);
    k_zero_f32<<<1, 256, 0, stream>>>(sums2, 2 * N2);
    k_zero_f32<<<2048, 256, 0, stream>>>(feat, (long long)U * 40);

    k_mask<<<(U + 255) / 256, 256, 0, stream>>>(cnt, pos, U);
    int nch = (U + 1023) / 1024;
    k_scan1<<<nch, 256, 0, stream>>>(pos, stmp, U);
    k_scan2<<<1, 256, 0, stream>>>(stmp, hdr, nch);
    k_scan3<<<(U + 255) / 256, 256, 0, stream>>>(pos, stmp, U);

    k_scatter<<<(Nv + 255) / 256, 256, 0, stream>>>(vf, vcrd, inv, feat, Nv);

    int totp = K1 * N1 + N1 * N2;
    k_pack_w<<<(totp + 255) / 256, 256, 0, stream>>>(bs_w1, bs_w2, w1p, w2p);

    k_stage_a<<<U, 64, 0, stream>>>(cnt, pos, feat, up_w1, up_b1, up_w2, up_b2,
                                    ca_w1, ca_w2, sa_c, flatA, U);

    k_gemm1<<<U16, 256, 0, stream>>>(flatA, w1p, z1row, sums1, hdr, U);
    k_bn1<<<Upad, 256, 0, stream>>>(z1row, sums1, bn1g, bn1b, z1p, hdr, U);
    k_gemm2<<<(Upad + 31) / 32, 256, 0, stream>>>(z1p, w2p, (float*)d_out, sums2, hdr, U);
    k_bn2<<<((size_t)U * 64 + 255) / 256, 256, 0, stream>>>((float*)d_out, sums2, bn2g, bn2b, hdr);
}
